// Segmentor_40724879900989
// MI455X (gfx1250) — matrix operand order repaired and checked by reading, not compiled, not run on hardware
//
#include <hip/hip_runtime.h>
#include <hip/hip_bf16.h>
#include <math.h>

// ---------------------------------------------------------------------------
// Problem constants (from reference)
// ---------------------------------------------------------------------------
#define BATCH 2
#define SEQ   400
#define INF_  80          // input feature dim I
#define HID   128         // H
#define FDIM  256         // F = 2H
#define NN    401         // N = S+1
#define G4H   512         // 4*H
#define KPAD  112         // 100 padded to 7 WMMA tiles of 16
#define WT_STRIDE 128     // Wt row stride in halves (256B rows -> 32B-aligned frags)
#define NT    26          // ceil(401/16)

typedef __attribute__((ext_vector_type(16))) _Float16 v16h;
typedef __attribute__((ext_vector_type(8)))  float    v8f;

__device__ __forceinline__ float preluf(float x, float a) {
    return (x >= 0.0f) ? x : a * x;
}
__device__ __forceinline__ float sigmf(float x) {
    return 1.0f / (1.0f + __expf(-x));
}

// ---------------------------------------------------------------------------
// Kernel 1: W1c (W1[:, :256]) -> f16, transposed, padded: Wt[f][kk], kk<128
// ---------------------------------------------------------------------------
__global__ void prep_wt_kernel(const float* __restrict__ W1, _Float16* __restrict__ Wt) {
    int idx = blockIdx.x * blockDim.x + threadIdx.x;     // 256*128
    if (idx >= FDIM * WT_STRIDE) return;
    int f  = idx / WT_STRIDE;
    int kk = idx % WT_STRIDE;
    float v = (kk < 100) ? W1[kk * 768 + f] : 0.0f;      // W1 is (100, 3F=768)
    Wt[f * WT_STRIDE + kk] = (_Float16)v;
}

// ---------------------------------------------------------------------------
// Kernel 2: xg[dir][b][t][g] = x[b][t] . Wi_dir[g] + b_dir[g]
// ---------------------------------------------------------------------------
__global__ void xproj_kernel(const float* __restrict__ x,
                             const float* __restrict__ Wi_f, const float* __restrict__ b_f,
                             const float* __restrict__ Wi_b, const float* __restrict__ b_b,
                             float* __restrict__ xg) {
    int idx = blockIdx.x * blockDim.x + threadIdx.x;     // 2*2*400*512
    if (idx >= 2 * BATCH * SEQ * G4H) return;
    int g   = idx % G4H;
    int t   = (idx / G4H) % SEQ;
    int b   = (idx / (G4H * SEQ)) % BATCH;
    int dir = idx / (G4H * SEQ * BATCH);
    const float* Wi = dir ? Wi_b : Wi_f;
    const float* bb = dir ? b_b  : b_f;
    const float* xr = x + (b * SEQ + t) * INF_;
    const float* wr = Wi + g * INF_;
    float acc = bb[g];
    #pragma unroll 4
    for (int i = 0; i < INF_; ++i) acc += xr[i] * wr[i];
    xg[idx] = acc;
}

// ---------------------------------------------------------------------------
// Kernel 3: LSTM recurrence. grid=2 (dir), block=512.
// rnn[b][t][dir*128+u]
// ---------------------------------------------------------------------------
__global__ void lstm_kernel(const float* __restrict__ xg,
                            const float* __restrict__ Wh_f,
                            const float* __restrict__ Wh_b,
                            float* __restrict__ rnn) {
    __shared__ float sh[BATCH][HID];
    __shared__ float sc[BATCH][HID];
    __shared__ float sg[BATCH][G4H];
    const int tid = threadIdx.x;
    const int dir = blockIdx.x;
    const float* Wh = dir ? Wh_b : Wh_f;

    if (tid < 2 * HID) {
        int b = tid >> 7, u = tid & 127;
        sh[b][u] = 0.0f; sc[b][u] = 0.0f;
    }
    __syncthreads();

    for (int s = 0; s < SEQ; ++s) {
        int t = dir ? (SEQ - 1 - s) : s;
        const float* wrow = Wh + tid * HID;
        float g0 = xg[((dir * 2 + 0) * SEQ + t) * G4H + tid];
        float g1 = xg[((dir * 2 + 1) * SEQ + t) * G4H + tid];
        #pragma unroll 4
        for (int u = 0; u < HID; ++u) {
            float w = wrow[u];
            g0 += w * sh[0][u];
            g1 += w * sh[1][u];
        }
        sg[0][tid] = g0;
        sg[1][tid] = g1;
        __syncthreads();
        if (tid < 2 * HID) {
            int b = tid >> 7, u = tid & 127;
            float gi = sg[b][u];
            float gf = sg[b][u + HID];
            float gg = sg[b][u + 2 * HID];
            float go = sg[b][u + 3 * HID];
            float c  = sigmf(gf) * sc[b][u] + sigmf(gi) * tanhf(gg);
            float h  = sigmf(go) * tanhf(c);
            sc[b][u] = c; sh[b][u] = h;
            rnn[(b * SEQ + t) * FDIM + dir * HID + u] = h;
        }
        __syncthreads();
    }
}

// ---------------------------------------------------------------------------
// Kernel 4: cum[b][n][f] = sum_{t<n} rnn[b][t][f]   (n = 0..400)
// ---------------------------------------------------------------------------
__global__ void cumsum_kernel(const float* __restrict__ rnn, float* __restrict__ cum) {
    int idx = blockIdx.x * blockDim.x + threadIdx.x;     // 512 = B*F
    if (idx >= BATCH * FDIM) return;
    int b = idx / FDIM, f = idx % FDIM;
    float run = 0.0f;
    cum[(b * NN + 0) * FDIM + f] = 0.0f;
    for (int t = 0; t < SEQ; ++t) {
        run += rnn[(b * SEQ + t) * FDIM + f];
        cum[(b * NN + t + 1) * FDIM + f] = run;
    }
}

// ---------------------------------------------------------------------------
// Kernel 5: pxd[b][n][k] = prelu(xp[b][n],a1) . W1d[k]; pxe same with W1e.
// Padded to KPAD with zeros.
// ---------------------------------------------------------------------------
__global__ void pxde_kernel(const float* __restrict__ rnn, const float* __restrict__ W1,
                            const float* __restrict__ a1p,
                            float* __restrict__ pxd, float* __restrict__ pxe) {
    int idx = blockIdx.x * blockDim.x + threadIdx.x;     // 2*401*112
    if (idx >= BATCH * NN * KPAD) return;
    int k = idx % KPAD;
    int n = (idx / KPAD) % NN;
    int b = idx / (KPAD * NN);
    if (k >= 100) { pxd[idx] = 0.0f; pxe[idx] = 0.0f; return; }
    float a1 = a1p[0];
    const float* wd = W1 + k * 768 + FDIM;       // W1d row
    const float* we = W1 + k * 768 + 2 * FDIM;   // W1e row
    float ad = 0.0f, ae = 0.0f;
    if (n > 0) {
        const float* xr = rnn + (b * SEQ + (n - 1)) * FDIM;
        #pragma unroll 4
        for (int f = 0; f < FDIM; ++f) {
            float px = preluf(xr[f], a1);
            ad += px * wd[f];
            ae += px * we[f];
        }
    }
    pxd[idx] = ad;
    pxe[idx] = ae;
}

// ---------------------------------------------------------------------------
// Kernel 6: fused pairwise scores via WMMA (the dominant GEMM).
// Per block: one (b, i-tile, j-tile) of 16x16 pairs. 8 waves, each owns
// 2 M-tiles (fixed i row, processed one after the other). K=256 in 8
// *non-unrolled* steps of v_wmma_f32_16x16x32_f16 (keeps live VGPRs ~100,
// no scratch spills), N=112 (7 tiles, unrolled so acc[] indexing is static).
// Epilogue folds bias/prelu/W2 reduction in-register (16-lane shuffles).
// ---------------------------------------------------------------------------
__global__ void __launch_bounds__(256)
scores_wmma_kernel(const float* __restrict__ cum,
                   const float* __restrict__ pxd, const float* __restrict__ pxe,
                   const _Float16* __restrict__ Wt,
                   const float* __restrict__ b1, const float* __restrict__ W2,
                   const float* __restrict__ a1p, const float* __restrict__ a2p,
                   const float* __restrict__ b2p,
                   float* __restrict__ scores) {
    __shared__ float sCumI[16][257];   // stride-257 pad: kills 16-way bank conflict
    __shared__ float sCumJ[16][257];
    __shared__ float sPd[16][KPAD];
    __shared__ float sPe[16][KPAD];
    __shared__ float w2s[KPAD];
    __shared__ float b1s[KPAD];

    const int tid  = threadIdx.x;
    int bid  = blockIdx.x;
    const int b   = bid / (NT * NT);
    const int rem = bid % (NT * NT);
    const int i0  = (rem / NT) * 16;
    const int j0  = (rem % NT) * 16;

    const float a1 = a1p[0];
    const float a2 = a2p[0];
    const float b2 = b2p[0];

    if (tid == 0) __builtin_prefetch(Wt, 0, 0);   // global_prefetch_b8: warm L2/WGP$

    // ---- stage tiles into LDS ----
    for (int idx = tid; idx < 16 * FDIM; idx += 256) {
        int r = idx >> 8, f = idx & 255;
        int gi = i0 + r, gj = j0 + r;
        sCumI[r][f] = (gi < NN) ? cum[(b * NN + gi) * FDIM + f] : 0.0f;
        sCumJ[r][f] = (gj < NN) ? cum[(b * NN + gj) * FDIM + f] : 0.0f;
    }
    for (int idx = tid; idx < 16 * KPAD; idx += 256) {
        int r = idx / KPAD, k = idx % KPAD;
        int gi = i0 + r, gj = j0 + r;
        sPd[r][k] = (gi < NN) ? pxd[(b * NN + gi) * KPAD + k] : 0.0f;
        sPe[r][k] = (gj < NN) ? pxe[(b * NN + gj) * KPAD + k] : 0.0f;
    }
    if (tid < KPAD) {
        w2s[tid] = (tid < 100) ? W2[tid] : 0.0f;
        b1s[tid] = (tid < 100) ? b1[tid] : 0.0f;
    }
    __syncthreads();

    const int wave = tid >> 5;
    const int lane = tid & 31;
    const int hsel = lane >> 4;        // half-wave select
    const int lcol = lane & 15;        // N column within tile / A row

    #pragma unroll 1
    for (int mti = 0; mti < 2; ++mti) {            // i_local = mt, 2 iters/wave
        const int mt = wave + mti * 8;

        v8f acc[7];
        #pragma unroll
        for (int n = 0; n < 7; ++n)
            #pragma unroll
            for (int r = 0; r < 8; ++r) acc[n][r] = 0.0f;

        #pragma unroll 1                           // keep live ranges small: no spills
        for (int s = 0; s < 8; ++s) {              // K step: f in [s*32, s*32+32)
            // A fragment: ISA 16-bit A 16x32 layout.
            // lane 0-15: M=lane, a[j] -> K = (j<8 ? j : j+8)
            // lane16-31: M=lane-16,a[j] -> K = (j<8 ? j+8 : j+16)
            v16h afrag;
            const int m  = lcol;                   // j_local row
            const int fb = s * 32 + (hsel << 3);
            #pragma unroll
            for (int j = 0; j < 16; ++j) {
                int f = fb + ((j < 8) ? j : j + 8);
                float d = sCumJ[m][f] - sCumI[mt][f];   // pc[i,j] = cum[j]-cum[i]
                afrag[j] = (_Float16)preluf(d, a1);
            }
            // B fragments: lane holds N-col = n*16+lcol, b[j] -> same K as a[j]:
            // b[j] = Wt[(fb + (j<8 ? j : j+8))*128 + n*16 + lcol]
            const _Float16* bcol = Wt + fb * WT_STRIDE + lcol;
            #pragma unroll
            for (int n = 0; n < 7; ++n) {
                union { v16h h; uint4 u[2]; } bb;
                #pragma unroll
                for (int j = 0; j < 16; ++j)
                    bb.h[j] = bcol[((j < 8) ? j : j + 8) * WT_STRIDE + n * 16];
                acc[n] = __builtin_amdgcn_wmma_f32_16x16x32_f16(
                    false, afrag, false, bb.h, (short)0, acc[n], false, false);
            }
        }

        // ---- epilogue: h -> prelu(a2) -> dot W2, reduced over k in-register
        float sval[8];
        #pragma unroll
        for (int r = 0; r < 8; ++r) sval[r] = 0.0f;

        #pragma unroll
        for (int n = 0; n < 7; ++n) {
            const int   kcol = n * 16 + lcol;
            const float w2v  = w2s[kcol];
            const float b1v  = b1s[kcol];
            const float pd   = sPd[mt][kcol];
            #pragma unroll
            for (int r = 0; r < 8; ++r) {
                int mrow = r + (hsel << 3);        // j_local
                float h = acc[n][r] + pd + sPe[mrow][kcol] + b1v;
                sval[r] += w2v * preluf(h, a2);
            }
        }
        #pragma unroll
        for (int r = 0; r < 8; ++r) {
            float v = sval[r];
            #pragma unroll
            for (int off = 8; off >= 1; off >>= 1)
                v += __shfl_xor(v, off, 16);       // reduce 16 k-columns
            if (lcol == 0) {
                int ig = i0 + mt;
                int jg = j0 + r + (hsel << 3);
                if (ig < NN && jg < NN)
                    scores[(b * NN + ig) * NN + jg] = v + b2;
            }
        }
    }
}

// ---------------------------------------------------------------------------
// Kernel 7: cls / bin heads. grid = B*S, block = 128.
// ---------------------------------------------------------------------------
__global__ void heads_kernel(const float* __restrict__ rnn,
                             const float* __restrict__ ac1p, const float* __restrict__ Wc1,
                             const float* __restrict__ bc1,  const float* __restrict__ ac2p,
                             const float* __restrict__ Wc2,  const float* __restrict__ bc2,
                             const float* __restrict__ ab1p, const float* __restrict__ Wb1,
                             const float* __restrict__ bb1,  const float* __restrict__ ab2p,
                             const float* __restrict__ Wb2,  const float* __restrict__ bb2,
                             float* __restrict__ cls_out, float* __restrict__ bin_out) {
    __shared__ float xrow[FDIM];
    __shared__ float h1[100];
    const int bt  = blockIdx.x;     // b*SEQ + t
    const int tid = threadIdx.x;
    for (int f = tid; f < FDIM; f += 128) xrow[f] = rnn[bt * FDIM + f];
    __syncthreads();

    // cls branch
    float ac1 = ac1p[0], ac2 = ac2p[0];
    if (tid < 100) {
        const float* w = Wc1 + tid * FDIM;
        float acc = bc1[tid];
        #pragma unroll 4
        for (int f = 0; f < FDIM; ++f) acc += preluf(xrow[f], ac1) * w[f];
        h1[tid] = preluf(acc, ac2);
    }
    __syncthreads();
    if (tid < 50) {
        const float* w = Wc2 + tid * 100;
        float acc = bc2[tid];
        #pragma unroll 4
        for (int q = 0; q < 100; ++q) acc += h1[q] * w[q];
        cls_out[bt * 50 + tid] = acc;
    }
    __syncthreads();

    // bin branch
    float ab1 = ab1p[0], ab2 = ab2p[0];
    if (tid < 100) {
        const float* w = Wb1 + tid * FDIM;
        float acc = bb1[tid];
        #pragma unroll 4
        for (int f = 0; f < FDIM; ++f) acc += preluf(xrow[f], ab1) * w[f];
        h1[tid] = preluf(acc, ab2);
    }
    __syncthreads();
    if (tid < 2) {
        const float* w = Wb2 + tid * 100;
        float acc = bb2[tid];
        #pragma unroll 4
        for (int q = 0; q < 100; ++q) acc += h1[q] * w[q];
        bin_out[bt * 2 + tid] = acc;
    }
}

// ---------------------------------------------------------------------------
// Kernel 8: segment DP (Viterbi over MAX_SEG=50 window). 1 block, 2 waves,
// one wave per batch row. Shuffle argmax with first-max tie-breaking.
// ---------------------------------------------------------------------------
__global__ void dp_kernel(const float* __restrict__ scores,
                          const int* __restrict__ lengths,
                          float* __restrict__ pred_out, int* __restrict__ prev_out) {
    __shared__ float sbest[BATCH][NN];
    const int tid  = threadIdx.x;
    const int b    = tid >> 5;
    const int lane = tid & 31;
    if (b >= BATCH) return;
    for (int n = lane; n < NN; n += 32) sbest[b][n] = 0.0f;
    if (lane == 0) prev_out[b * NN + 0] = 0;

    for (int i = 1; i < NN; ++i) {
        int lo = i - 50; if (lo < 0) lo = 0;
        float m  = -INFINITY;
        int   kb = 0x7fffffff;
        for (int k = lo + lane; k < i; k += 32) {
            float v = sbest[b][k] + scores[(b * NN + k) * NN + i];
            if (v > m || (v == m && k < kb)) { m = v; kb = k; }
        }
        #pragma unroll
        for (int off = 16; off >= 1; off >>= 1) {
            float om = __shfl_xor(m, off, 32);
            int   ok = __shfl_xor(kb, off, 32);
            if (om > m || (om == m && ok < kb)) { m = om; kb = ok; }
        }
        if (lane == 0) {
            sbest[b][i] = m;
            prev_out[b * NN + i] = kb;
        }
    }
    if (lane == 0) {
        int L = lengths[b];
        if (L < 0) L = 0; if (L >= NN) L = NN - 1;
        pred_out[b] = sbest[b][L];
    }
}

// ---------------------------------------------------------------------------
// Host launcher
// ---------------------------------------------------------------------------
extern "C" void kernel_launch(void* const* d_in, const int* in_sizes, int n_in,
                              void* d_out, int out_size, void* d_ws, size_t ws_size,
                              hipStream_t stream) {
    (void)in_sizes; (void)n_in; (void)out_size; (void)ws_size;

    const float* x    = (const float*)d_in[0];
    const int*   len  = (const int*)  d_in[1];
    const float* Wi_f = (const float*)d_in[2];
    const float* Wh_f = (const float*)d_in[3];
    const float* b_f  = (const float*)d_in[4];
    const float* Wi_b = (const float*)d_in[5];
    const float* Wh_b = (const float*)d_in[6];
    const float* b_b  = (const float*)d_in[7];
    const float* a1   = (const float*)d_in[8];
    const float* W1   = (const float*)d_in[9];
    const float* b1   = (const float*)d_in[10];
    const float* a2   = (const float*)d_in[11];
    const float* W2   = (const float*)d_in[12];
    const float* b2   = (const float*)d_in[13];
    const float* ac1  = (const float*)d_in[14];
    const float* Wc1  = (const float*)d_in[15];
    const float* bc1  = (const float*)d_in[16];
    const float* ac2  = (const float*)d_in[17];
    const float* Wc2  = (const float*)d_in[18];
    const float* bc2  = (const float*)d_in[19];
    const float* ab1  = (const float*)d_in[20];
    const float* Wb1  = (const float*)d_in[21];
    const float* bb1  = (const float*)d_in[22];
    const float* ab2  = (const float*)d_in[23];
    const float* Wb2  = (const float*)d_in[24];
    const float* bb2  = (const float*)d_in[25];

    // workspace layout (bytes); cum/pxd/pxe reuse the xg region (dead after LSTM)
    char* ws = (char*)d_ws;
    float*    xg   = (float*)   (ws + 0);          // 2*2*400*512*4 = 3,276,800
    float*    cum  = (float*)   (ws + 0);          // 2*401*256*4   =   821,248 (after lstm)
    float*    pxd  = (float*)   (ws + 1048576);    // 2*401*112*4   =   359,296
    float*    pxe  = (float*)   (ws + 1441792);
    float*    rnn  = (float*)   (ws + 3276800);    // 2*400*256*4   =   819,200
    _Float16* Wt   = (_Float16*)(ws + 4096000);    // 256*128*2     =    65,536

    // output layout (floats, concatenated in return order)
    float* out      = (float*)d_out;
    float* scores   = out;                                   // 2*401*401 = 321602
    float* cls_out  = out + BATCH * NN * NN;                 // 2*400*50  =  40000
    float* bin_out  = cls_out + BATCH * SEQ * 50;            // 2*400*2   =   1600
    float* pred_out = bin_out + BATCH * SEQ * 2;             // 2
    int*   prev_out = (int*)(pred_out + BATCH);              // 2*401 int32

    // 1. W1c -> f16 transposed/padded
    prep_wt_kernel<<<(FDIM * WT_STRIDE + 255) / 256, 256, 0, stream>>>(W1, Wt);
    // 2. x @ Wi.T + b, both directions
    xproj_kernel<<<(2 * BATCH * SEQ * G4H + 255) / 256, 256, 0, stream>>>(
        x, Wi_f, b_f, Wi_b, b_b, xg);
    // 3. LSTM recurrence (one block per direction)
    lstm_kernel<<<2, G4H, 0, stream>>>(xg, Wh_f, Wh_b, rnn);
    // 4. cumsum of padded rnn_out
    cumsum_kernel<<<(BATCH * FDIM + 255) / 256, 256, 0, stream>>>(rnn, cum);
    // 5. per-row projections pxd/pxe (k-padded)
    pxde_kernel<<<(BATCH * NN * KPAD + 255) / 256, 256, 0, stream>>>(
        rnn, W1, a1, pxd, pxe);
    // 6. fused pairwise score GEMM (WMMA)
    scores_wmma_kernel<<<BATCH * NT * NT, 256, 0, stream>>>(
        cum, pxd, pxe, Wt, b1, W2, a1, a2, b2, scores);
    // 7. cls / bin heads
    heads_kernel<<<BATCH * SEQ, 128, 0, stream>>>(
        rnn, ac1, Wc1, bc1, ac2, Wc2, bc2, ab1, Wb1, bb1, ab2, Wb2, bb2,
        cls_out, bin_out);
    // 8. segment DP + pred_scores + prev
    dp_kernel<<<1, 64, 0, stream>>>(scores, len, pred_out, prev_out);
}
